// Fast_RCNN_21672404976101
// MI455X (gfx1250) — matrix operand order repaired and checked by reading, not compiled, not run on hardware
//
#include <hip/hip_runtime.h>
#include <hip/hip_bf16.h>

typedef __attribute__((ext_vector_type(16))) __bf16          bf16x16;
typedef __attribute__((ext_vector_type(8)))  float           f32x8;
typedef __attribute__((ext_vector_type(8)))  unsigned short  u16x8;

union FragU { u16x8 h[2]; bf16x16 v; };
union PackU { unsigned short a[8]; u16x8 v; };

__device__ __forceinline__ unsigned short f2bf(float f) {
  unsigned int u = __builtin_bit_cast(unsigned int, f);
  unsigned int r = u + 0x7FFFu + ((u >> 16) & 1u);   // round-to-nearest-even
  return (unsigned short)(r >> 16);
}
__device__ __forceinline__ float bf2f(unsigned short h) {
  unsigned int u = ((unsigned int)h) << 16;
  return __builtin_bit_cast(float, u);
}

#define BM 64
#define BN 256
#define BK 32
#define LDK 40   // padded LDS row stride in ushorts (80B -> conflict-free b128)

// ---------------------------------------------------------------------------
// Implicit-GEMM 3x3 SAME conv:  Y[b,co,h,w] = relu(s[co]*sum + b[co]),  bf16 io
// M = Cout (block 64), N = B*H*W (block 256), K = Cin*9 (step 32)
// 8 waves: wmg = wave&1 -> 32 M rows (2 tiles), wng = wave>>1 -> 64 N cols (4 tiles)
// ---------------------------------------------------------------------------
__global__ __launch_bounds__(256)
void conv_wmma(const unsigned short* __restrict__ X, const float* __restrict__ Wt,
               const float* __restrict__ Sc, const float* __restrict__ Bi,
               unsigned short* __restrict__ Y,
               int Cin, int H, int Wd, int Cout, int NP, int Keff)
{
  __shared__ unsigned short lA[BM * LDK];
  __shared__ unsigned short lB[BN * LDK];

  const int tid  = threadIdx.x;
  const int lane = tid & 31;
  const int wave = tid >> 5;
  const int wmg  = wave & 1;     // M group: rows wmg*32..+31
  const int wng  = wave >> 1;    // N group: cols wng*64..+63
  const int n0   = blockIdx.x * BN;
  const int co0  = blockIdx.y * BM;
  const int HW   = H * Wd;

  f32x8 acc[2][4];
  const f32x8 zero = {0.f,0.f,0.f,0.f,0.f,0.f,0.f,0.f};
  #pragma unroll
  for (int mi = 0; mi < 2; ++mi)
    #pragma unroll
    for (int t = 0; t < 4; ++t) acc[mi][t] = zero;

  // ---- hoisted per-thread spatial decomposition (B writer owns n = tid) ----
  const int  p    = n0 + tid;
  const bool inim = (p < NP);
  int b = 0, h = 0, w = 0;
  size_t base0 = 0;
  if (inim) {
    b  = p / HW;
    int rp = p - b * HW;
    h  = rp / Wd;
    w  = rp - h * Wd;
    base0 = (size_t)b * Cin * HW;
  }

  // ---- hoisted A-writer indices (thread -> row m, 8-wide k chunk) ----------
  const int am  = tid >> 2;
  const int akq = (tid & 3) << 3;

  const int Ktiles = (Keff + BK - 1) / BK;
  for (int kt = 0; kt < Ktiles; ++kt) {
    const int k0 = kt * BK;
    __syncthreads();

    // ---- A tile: weights [64 x 32] fp32 -> bf16, one b128 store/thread -----
    {
      const float* src = Wt + (size_t)(co0 + am) * Keff + (k0 + akq);
      PackU up;
      #pragma unroll
      for (int u = 0; u < 8; ++u) {
        float wv = (k0 + akq + u < Keff) ? src[u] : 0.f;
        up.a[u] = f2bf(wv);
      }
      *(u16x8*)(&lA[am * LDK + akq]) = up.v;
    }

    // ---- B tile: im2col gather, one n per thread, 8x ds_store_b64 ----------
    #pragma unroll
    for (int kg = 0; kg < 8; ++kg) {
      ushort4 d;
      #pragma unroll
      for (int u = 0; u < 4; ++u) {
        const int k  = (kg << 2) + u;
        const int kk = k0 + k;
        unsigned short val = 0;
        if (inim && kk < Keff) {
          int ci = kk / 9;             // constant-divisor: mul+shift
          int r9 = kk - ci * 9;
          int kh = r9 / 3;
          int kw = r9 - kh * 3;
          int hh = h + kh - 1;
          int ww = w + kw - 1;
          if ((unsigned)hh < (unsigned)H && (unsigned)ww < (unsigned)Wd)
            val = X[base0 + (size_t)ci * HW + (size_t)hh * Wd + ww];
        }
        ((unsigned short*)&d)[u] = val;
      }
      *(ushort4*)(&lB[tid * LDK + (kg << 2)]) = d;
    }

    if (kt + 1 < Ktiles)  // stream next weight tile toward L2 (global_prefetch_b8)
      __builtin_prefetch(Wt + (size_t)(co0 + (tid & 63)) * Keff + (k0 + BK), 0, 1);

    __syncthreads();

    // ---- fragments + WMMA: 2 A frags, 4 B frags, 8 WMMAs per wave ----------
    FragU fa[2];
    const int akb = (lane >> 4) << 3;     // lanes>=16: K base 8 / 24
    #pragma unroll
    for (int mi = 0; mi < 2; ++mi) {
      const int ar = ((wmg * 2 + mi) << 4) + (lane & 15);
      fa[mi].h[0] = *(const u16x8*)(&lA[ar * LDK + akb]);
      fa[mi].h[1] = *(const u16x8*)(&lA[ar * LDK + akb + 16]);
    }
    const int bkb = (lane >> 4) << 3;     // lanes>=16: K base 8 / 24
    #pragma unroll
    for (int t = 0; t < 4; ++t) {
      FragU fb;
      const int bn = ((wng * 4 + t) << 4) + (lane & 15);
      fb.h[0] = *(const u16x8*)(&lB[bn * LDK + bkb]);
      fb.h[1] = *(const u16x8*)(&lB[bn * LDK + bkb + 16]);
      #pragma unroll
      for (int mi = 0; mi < 2; ++mi)
        acc[mi][t] = __builtin_amdgcn_wmma_f32_16x16x32_bf16(
                       false, fa[mi].v, false, fb.v, (short)0, acc[mi][t], false, false);
    }
  }

  // ---- epilogue: scale/bias/relu, bf16 store -------------------------------
  const int hiL  = lane >> 4;
  const int lcol = lane & 15;
  #pragma unroll
  for (int t = 0; t < 4; ++t) {
    int pp = n0 + ((wng * 4 + t) << 4) + lcol;
    if (pp < NP) {
      int bb = pp / HW;
      int rp = pp - bb * HW;
      int hh = rp / Wd;
      int ww = rp - hh * Wd;
      #pragma unroll
      for (int mi = 0; mi < 2; ++mi) {
        #pragma unroll
        for (int v = 0; v < 8; ++v) {
          int co = co0 + (wmg << 5) + (mi << 4) + (hiL << 3) + v;
          float y = acc[mi][t][v] * Sc[co] + Bi[co];
          y = y > 0.f ? y : 0.f;
          Y[(size_t)((bb * Cout + co) * H + hh) * Wd + ww] = f2bf(y);
        }
      }
    }
  }
}

// ---------------------------------------------------------------------------
// FC GEMM:  Y[m,n] = act(sum_k X[m,k]*W[n,k] + bias[n]);  X bf16, W fp32
// Same tiling: BM=64, BN=256, BK=32, 8 WMMAs / wave / K-step
// ---------------------------------------------------------------------------
__global__ __launch_bounds__(256)
void fc_wmma(const unsigned short* __restrict__ X, const float* __restrict__ Wt,
             const float* __restrict__ Bias, unsigned short* __restrict__ Y,
             int M, int N, int Kd, int doRelu)
{
  __shared__ unsigned short lA[BM * LDK];
  __shared__ unsigned short lB[BN * LDK];

  const int tid  = threadIdx.x;
  const int lane = tid & 31;
  const int wave = tid >> 5;
  const int wmg  = wave & 1;
  const int wng  = wave >> 1;
  const int m0   = blockIdx.y * BM;
  const int nb   = blockIdx.x * BN;

  f32x8 acc[2][4];
  const f32x8 zero = {0.f,0.f,0.f,0.f,0.f,0.f,0.f,0.f};
  #pragma unroll
  for (int mi = 0; mi < 2; ++mi)
    #pragma unroll
    for (int t = 0; t < 4; ++t) acc[mi][t] = zero;

  const int am  = tid >> 2;
  const int akq = (tid & 3) << 3;

  const int Ktiles = Kd >> 5;            // Kd is a multiple of 32 here
  for (int kt = 0; kt < Ktiles; ++kt) {
    const int k0 = kt << 5;
    __syncthreads();

    // ---- A tile: bf16 activation rows, one aligned b128 load+store ---------
    {
      u16x8 d = {0,0,0,0,0,0,0,0};
      if (m0 + am < M)
        d = *(const u16x8*)(&X[(size_t)(m0 + am) * Kd + k0 + akq]);
      *(u16x8*)(&lA[am * LDK + akq]) = d;
    }

    // ---- B tile: one weight row per thread, 8 float4 -> 4 b128 stores ------
    {
      const float* src = Wt + (size_t)(nb + tid) * Kd + k0;
      #pragma unroll
      for (int g = 0; g < 4; ++g) {
        const float4 w0 = *(const float4*)(src + (g << 3));
        const float4 w1 = *(const float4*)(src + (g << 3) + 4);
        PackU up;
        up.a[0] = f2bf(w0.x); up.a[1] = f2bf(w0.y);
        up.a[2] = f2bf(w0.z); up.a[3] = f2bf(w0.w);
        up.a[4] = f2bf(w1.x); up.a[5] = f2bf(w1.y);
        up.a[6] = f2bf(w1.z); up.a[7] = f2bf(w1.w);
        *(u16x8*)(&lB[tid * LDK + (g << 3)]) = up.v;
      }
    }

    if (kt + 1 < Ktiles)
      __builtin_prefetch(Wt + (size_t)(nb + tid) * Kd + (k0 + BK), 0, 1);

    __syncthreads();

    FragU fa[2];
    const int akb = (lane >> 4) << 3;
    #pragma unroll
    for (int mi = 0; mi < 2; ++mi) {
      const int ar = ((wmg * 2 + mi) << 4) + (lane & 15);
      fa[mi].h[0] = *(const u16x8*)(&lA[ar * LDK + akb]);
      fa[mi].h[1] = *(const u16x8*)(&lA[ar * LDK + akb + 16]);
    }
    const int bkb = (lane >> 4) << 3;
    #pragma unroll
    for (int t = 0; t < 4; ++t) {
      FragU fb;
      const int bn = ((wng * 4 + t) << 4) + (lane & 15);
      fb.h[0] = *(const u16x8*)(&lB[bn * LDK + bkb]);
      fb.h[1] = *(const u16x8*)(&lB[bn * LDK + bkb + 16]);
      #pragma unroll
      for (int mi = 0; mi < 2; ++mi)
        acc[mi][t] = __builtin_amdgcn_wmma_f32_16x16x32_bf16(
                       false, fa[mi].v, false, fb.v, (short)0, acc[mi][t], false, false);
    }
  }

  const int hiL  = lane >> 4;
  const int lcol = lane & 15;
  #pragma unroll
  for (int t = 0; t < 4; ++t) {
    int n = nb + ((wng * 4 + t) << 4) + lcol;
    float bia = Bias[n];
    #pragma unroll
    for (int mi = 0; mi < 2; ++mi) {
      #pragma unroll
      for (int v = 0; v < 8; ++v) {
        int m = m0 + (wmg << 5) + (mi << 4) + (hiL << 3) + v;
        if (m < M) {
          float y = acc[mi][t][v] + bia;
          if (doRelu) y = y > 0.f ? y : 0.f;
          Y[(size_t)m * N + n] = f2bf(y);
        }
      }
    }
  }
}

// ---------------------------------------------------------------------------
// Small helper kernels
// ---------------------------------------------------------------------------
__global__ void cvt_bf16(const float* __restrict__ in, unsigned short* __restrict__ out, int n) {
  int i = blockIdx.x * 256 + threadIdx.x;
  if (i < n) out[i] = f2bf(in[i]);
}

__global__ void maxpool2(const unsigned short* __restrict__ in, unsigned short* __restrict__ out,
                         int BC, int H, int W)
{
  int OH = H >> 1, OW = W >> 1;
  int tot = BC * OH * OW;
  int idx = blockIdx.x * 256 + threadIdx.x;
  if (idx >= tot) return;
  int bc = idx / (OH * OW);
  int r  = idx - bc * OH * OW;
  int oh = r / OW, ow = r - oh * OW;
  const unsigned short* p = in + (size_t)bc * H * W + (oh * 2) * W + ow * 2;
  float mx = fmaxf(fmaxf(bf2f(p[0]), bf2f(p[1])), fmaxf(bf2f(p[W]), bf2f(p[W + 1])));
  out[idx] = f2bf(mx);
}

// ROI max-pool, replicating the reference (x-coords -> h, y-coords -> w,
// jnp.round == round-half-even == rintf). Feature map: [2,512,14,14] bf16.
__global__ void roipool(const unsigned short* __restrict__ F, const int* __restrict__ R,
                        unsigned short* __restrict__ Xo)
{
  int idx = blockIdx.x * 256 + threadIdx.x;
  const int TOT = 128 * 512 * 49;
  if (idx >= TOT) return;
  int m  = idx / (512 * 49);
  int rm = idx - m * 512 * 49;
  int c  = rm / 49;
  int ij = rm - c * 49;
  int i  = ij / 7, j = ij - i * 7;
  int b  = m >> 6;
  const int* rg = R + m * 4;
  const float ratio = 14.0f / 224.0f;
  int x0 = (int)rintf((float)rg[0] * ratio);
  int y0 = (int)rintf((float)rg[1] * ratio);
  int x1 = (int)rintf((float)rg[2] * ratio);
  int y1 = (int)rintf((float)rg[3] * ratio);
  int lx = x1 - x0 + 1, ly = y1 - y0 + 1;
  int hs = x0 + (i * lx) / 7,  he = x0 + ((i + 1) * lx + 6) / 7;
  int ws = y0 + (j * ly) / 7,  we = y0 + ((j + 1) * ly + 6) / 7;
  hs = hs > 0 ? hs : 0;  he = he < 14 ? he : 14;
  ws = ws > 0 ? ws : 0;  we = we < 14 ? we : 14;
  float mx = -INFINITY;
  const unsigned short* fp = F + (size_t)(b * 512 + c) * 196;
  for (int h = hs; h < he; ++h)
    for (int w = ws; w < we; ++w)
      mx = fmaxf(mx, bf2f(fp[h * 14 + w]));
  Xo[(size_t)m * 25088 + c * 49 + ij] = f2bf(mx);
}

// cls (2x4096) + bbox (8x4096) heads; fp32 output, concatenated.
__global__ void heads(const unsigned short* __restrict__ X,
                      const float* __restrict__ CW, const float* __restrict__ CB,
                      const float* __restrict__ BW, const float* __restrict__ BB,
                      float* __restrict__ out)
{
  int g = blockIdx.x * 256 + threadIdx.x;
  if (g < 256) {
    int m = g >> 1, n = g & 1;
    const float* wv = CW + n * 4096;
    const unsigned short* xv = X + (size_t)m * 4096;
    float s = CB[n];
    for (int k = 0; k < 4096; ++k) s += bf2f(xv[k]) * wv[k];
    out[g] = s;
  } else if (g < 1280) {
    int g2 = g - 256;
    int m = g2 >> 3, n = g2 & 7;
    const float* wv = BW + n * 4096;
    const unsigned short* xv = X + (size_t)m * 4096;
    float s = BB[n];
    for (int k = 0; k < 4096; ++k) s += bf2f(xv[k]) * wv[k];
    out[256 + g2] = s;
  }
}

// ---------------------------------------------------------------------------
extern "C" void kernel_launch(void* const* d_in, const int* in_sizes, int n_in,
                              void* d_out, int out_size, void* d_ws, size_t ws_size,
                              hipStream_t stream)
{
  (void)in_sizes; (void)out_size; (void)ws_size;
  if (n_in < 49) return;

  const float* inp     = (const float*)d_in[0];
  const int*   regions = (const int*)d_in[1];
  const float* cw[13]; const float* cs[13]; const float* cb[13];
  for (int i = 0; i < 13; ++i) {
    cw[i] = (const float*)d_in[2 + 3 * i];
    cs[i] = (const float*)d_in[3 + 3 * i];
    cb[i] = (const float*)d_in[4 + 3 * i];
  }
  const float* fc1w = (const float*)d_in[41];
  const float* fc1b = (const float*)d_in[42];
  const float* fc2w = (const float*)d_in[43];
  const float* fc2b = (const float*)d_in[44];
  const float* clsw = (const float*)d_in[45];
  const float* clsb = (const float*)d_in[46];
  const float* bbw  = (const float*)d_in[47];
  const float* bbb  = (const float*)d_in[48];

  char* ws = (char*)d_ws;
  const size_t ACT = 12845056;                     // 2*64*224*224 * 2B (max act)
  unsigned short* A0 = (unsigned short*)(ws);
  unsigned short* A1 = (unsigned short*)(ws + ACT);
  unsigned short* XP = (unsigned short*)(ws + 2 * ACT);                      // 128x25088
  unsigned short* F1 = (unsigned short*)(ws + 2 * ACT + 6422528);            // 128x4096
  unsigned short* F2 = (unsigned short*)(ws + 2 * ACT + 6422528 + 1048576);  // 128x4096

  { int n = 2 * 3 * 224 * 224;
    cvt_bf16<<<(n + 255) / 256, 256, 0, stream>>>(inp, A0, n); }

  unsigned short* cur = A0; unsigned short* nxt = A1;
  int H = 224, Cin = 3, ci = 0;
  const int cfg[17] = {64,64,-1,128,128,-1,256,256,256,-1,512,512,512,-1,512,512,512};
  for (int st = 0; st < 17; ++st) {
    int v = cfg[st];
    if (v < 0) {
      int OH = H / 2;
      int n = 2 * Cin * OH * OH;
      maxpool2<<<(n + 255) / 256, 256, 0, stream>>>(cur, nxt, 2 * Cin, H, H);
      H = OH;
    } else {
      int NP   = 2 * H * H;
      int Keff = Cin * 9;
      dim3 g((NP + BN - 1) / BN, v / BM);
      conv_wmma<<<g, 256, 0, stream>>>(cur, cw[ci], cs[ci], cb[ci], nxt,
                                       Cin, H, H, v, NP, Keff);
      Cin = v; ++ci;
    }
    unsigned short* t = cur; cur = nxt; nxt = t;
  }

  roipool<<<(128 * 512 * 49 + 255) / 256, 256, 0, stream>>>(cur, regions, XP);

  { dim3 g(4096 / BN, (128 + BM - 1) / BM);
    fc_wmma<<<g, 256, 0, stream>>>(XP, fc1w, fc1b, F1, 128, 4096, 25088, 1); }
  { dim3 g(4096 / BN, (128 + BM - 1) / BM);
    fc_wmma<<<g, 256, 0, stream>>>(F1, fc2w, fc2b, F2, 128, 4096, 4096, 1); }

  heads<<<5, 256, 0, stream>>>(F2, clsw, clsb, bbw, bbb, (float*)d_out);
}
